// EncoderBlock_16226386444669
// MI455X (gfx1250) — matrix operand order repaired and checked by reading, not compiled, not run on hardware
//
#include <hip/hip_runtime.h>
#include <hip/hip_bf16.h>
#include <math.h>

// ---------------------------------------------------------------------------
// GPT-2-style encoder block for MI455X (gfx1250), bf16 WMMA everywhere.
// Software-pipelined GEMM inner loop (load K-step k+1 under the WMMAs of k).
// ---------------------------------------------------------------------------

typedef __bf16 bf16_t;
typedef __bf16 v16bf __attribute__((ext_vector_type(16)));
typedef __bf16 v8bf  __attribute__((ext_vector_type(8)));
typedef float  v8f   __attribute__((ext_vector_type(8)));
typedef float  v4f   __attribute__((ext_vector_type(4)));

#define D_MODEL 768
#define N_HEADS 12
#define DK      64
#define D_FFN   3072
#define BATCH   16
#define SEQ     1024
#define TOK     (BATCH * SEQ)   // 16384 tokens

__device__ inline v16bf cat8(v8bf lo, v8bf hi) {
    v16bf a;
#pragma unroll
    for (int e = 0; e < 8; ++e) { a[e] = lo[e]; a[8 + e] = hi[e]; }
    return a;
}

__device__ inline v16bf load_a_frag(const bf16_t* p, int hh) {
    // A-fragment: lane row r, elements 0..15 = K[hh*16..hh*16+15]
    return cat8(*(const v8bf*)(p + hh * 16), *(const v8bf*)(p + hh * 16 + 8));
}

__device__ inline v8f wmma_bf16(v16bf a, v16bf b, v8f c) {
    // D = A(16x32) * B(32x16) + C, fp32 accumulate
    return __builtin_amdgcn_wmma_f32_16x16x32_bf16(
        /*neg_a=*/false, a, /*neg_b=*/false, b,
        /*c_mod=*/(short)0, c, /*reuse_a=*/false, /*reuse_b=*/false);
}

// ---------------------------------------------------------------------------
// Pack fp32 row-major W[K][N] into bf16 WMMA B-fragment order:
//   Bp[((kt*(N/16)+nt)*32 + lane)*16 + i] = W[kt*32 + (lane/16)*16 + i][nt*16 + lane%16]
// so a GEMM lane's whole B fragment is one contiguous 32-byte load.
// ---------------------------------------------------------------------------
__global__ __launch_bounds__(256)
void pack_w(const float* __restrict__ W, bf16_t* __restrict__ P, int K, int N) {
    size_t t = (size_t)blockIdx.x * 256 + threadIdx.x;
    size_t total = (size_t)K * N;
    if (t >= total) return;
    int i    = (int)(t & 15);
    int lane = (int)((t >> 4) & 31);
    size_t tile = t >> 9;
    int n16 = N >> 4;
    int nt = (int)(tile % n16);
    int kt = (int)(tile / n16);
    int row = kt * 32 + (lane >> 4) * 16 + i;
    int col = nt * 16 + (lane & 15);
    P[t] = (bf16_t)W[(size_t)row * N + col];
}

// ---------------------------------------------------------------------------
// LayerNorm over D_MODEL, one wave32 per token, output bf16.
// ---------------------------------------------------------------------------
__global__ __launch_bounds__(256)
void layernorm_bf16(const float* __restrict__ X, const float* __restrict__ g,
                    const float* __restrict__ b, bf16_t* __restrict__ Y, int T) {
    int wave = threadIdx.x >> 5, lane = threadIdx.x & 31;
    int t = blockIdx.x * 8 + wave;
    if (t >= T) return;
    const float* x = X + (size_t)t * D_MODEL;
    float s = 0.f, s2 = 0.f;
    for (int j = lane; j < D_MODEL; j += 32) { float v = x[j]; s += v; s2 += v * v; }
#pragma unroll
    for (int m = 16; m > 0; m >>= 1) {
        s  += __shfl_xor(s,  m, 32);
        s2 += __shfl_xor(s2, m, 32);
    }
    float mu  = s * (1.f / D_MODEL);
    float var = s2 * (1.f / D_MODEL) - mu * mu;
    float inv = rsqrtf(var + 1e-5f);
    bf16_t* y = Y + (size_t)t * D_MODEL;
    for (int j = lane; j < D_MODEL; j += 32)
        y[j] = (bf16_t)((x[j] - mu) * inv * g[j] + b[j]);
}

// ---------------------------------------------------------------------------
// V[t][h*64+d]  ->  VT[b][h][d][s]  (bf16) so P@V B-fragments are contiguous.
// ---------------------------------------------------------------------------
__global__ __launch_bounds__(256)
void transpose_v(const bf16_t* __restrict__ V, bf16_t* __restrict__ VT) {
    size_t t = (size_t)blockIdx.x * 256 + threadIdx.x;
    if (t >= (size_t)TOK * D_MODEL) return;
    int s = (int)(t % SEQ); size_t rest = t / SEQ;
    int d = (int)(rest % DK); rest /= DK;
    int h = (int)(rest % N_HEADS);
    int bz = (int)(rest / N_HEADS);
    VT[t] = V[((size_t)(bz * SEQ + s)) * D_MODEL + h * DK + d];
}

// ---------------------------------------------------------------------------
// bf16 GEMM: out[M x N] = act(A[M x K] @ Bpacked + bias [+ residual])
// One wave computes a 16x64 tile (4 WMMA accumulators), 8 waves per block.
// Inner loop is software-pipelined one K-step ahead; B addressing uses a
// single pointer + constant immediate offsets (no per-fragment SALU).
// ---------------------------------------------------------------------------
__global__ __launch_bounds__(256)
void gemm_bf16_wmma(const bf16_t* __restrict__ A,
                    const bf16_t* __restrict__ Bp,
                    const float*  __restrict__ bias,
                    const float*  __restrict__ residual,
                    float*        __restrict__ outF,
                    bf16_t*       __restrict__ outB,
                    int M, int N, int K, int gelu) {
    const int lane = threadIdx.x & 31;
    const int wave = threadIdx.x >> 5;
    const int mt = blockIdx.x * 8 + wave;   // 16-row tile index
    const int nb = blockIdx.y;              // 64-col block
    const int m0 = mt * 16;
    const int r  = lane & 15;
    const int hh = lane >> 4;
    const int n16 = N >> 4;

    v8f acc[4];
#pragma unroll
    for (int nt = 0; nt < 4; ++nt)
#pragma unroll
        for (int i = 0; i < 8; ++i) acc[nt][i] = 0.f;

    const bf16_t* aptr = A + (size_t)(m0 + r) * K;
    const bf16_t* bptr = Bp + ((size_t)(nb * 4) * 32 + lane) * 16;
    const size_t  bstep = (size_t)n16 * 512;   // elements per 32-wide K-step
    const int kt_end = K >> 5;

    // prologue: fragments for K-step 0
    v16bf a  = load_a_frag(aptr, hh);
    v16bf b0 = *(const v16bf*)(bptr);
    v16bf b1 = *(const v16bf*)(bptr + 512);
    v16bf b2 = *(const v16bf*)(bptr + 1024);
    v16bf b3 = *(const v16bf*)(bptr + 1536);

    for (int kt = 0; kt < kt_end - 1; ++kt) {
        aptr += 32; bptr += bstep;
        // issue next K-step's loads before consuming current fragments
        v16bf an = load_a_frag(aptr, hh);
        v16bf c0 = *(const v16bf*)(bptr);
        v16bf c1 = *(const v16bf*)(bptr + 512);
        v16bf c2 = *(const v16bf*)(bptr + 1024);
        v16bf c3 = *(const v16bf*)(bptr + 1536);
        __builtin_prefetch(aptr + 32 + hh * 8, 0, 3);   // near-scope prefetch
        acc[0] = wmma_bf16(a, b0, acc[0]);
        acc[1] = wmma_bf16(a, b1, acc[1]);
        acc[2] = wmma_bf16(a, b2, acc[2]);
        acc[3] = wmma_bf16(a, b3, acc[3]);
        a = an; b0 = c0; b1 = c1; b2 = c2; b3 = c3;
    }
    acc[0] = wmma_bf16(a, b0, acc[0]);
    acc[1] = wmma_bf16(a, b1, acc[1]);
    acc[2] = wmma_bf16(a, b2, acc[2]);
    acc[3] = wmma_bf16(a, b3, acc[3]);

#pragma unroll
    for (int nt = 0; nt < 4; ++nt) {
        const int col = nb * 64 + nt * 16 + r;
        const float bsv = bias ? bias[col] : 0.f;
#pragma unroll
        for (int i = 0; i < 8; ++i) {
            const int row = m0 + hh * 8 + i;
            float v = acc[nt][i] + bsv;
            if (residual) v += residual[(size_t)row * N + col];
            if (gelu) v = 0.5f * v * (1.0f + erff(v * 0.70710678118654752f));
            if (outF) outF[(size_t)row * N + col] = v;
            else      outB[(size_t)row * N + col] = (bf16_t)v;
        }
    }
}

// ---------------------------------------------------------------------------
// Flash attention: one wave per 16 query rows of one (batch, head).
// QK^T and P@V via bf16 WMMA; online softmax via per-wave LDS tile with
// explicit s_wait_dscnt fences.  V fragments are loaded right after the
// score WMMAs so the global loads overlap the LDS softmax round-trip.
// ---------------------------------------------------------------------------
__global__ __launch_bounds__(256)
void flash_attn(const bf16_t* __restrict__ Q, const bf16_t* __restrict__ Km,
                const bf16_t* __restrict__ VT, bf16_t* __restrict__ O) {
    __shared__ float  sS[8][16 * 32];  // raw scores (row-major 16x32)
    __shared__ bf16_t sP[8][16 * 32];  // probabilities (row-major, A reads)
    __shared__ float  sF[8][16];       // per-row broadcast (factor / 1/l)

    const int lane = threadIdx.x & 31;
    const int wave = threadIdx.x >> 5;
    const int qt = blockIdx.x * 8 + wave;       // 16-query tile (0..63)
    const int bh = blockIdx.y;                  // batch*N_HEADS + head
    const int bz = bh / N_HEADS, hd = bh % N_HEADS;
    const int r  = lane & 15;
    const int hh = lane >> 4;
    const float scale = 0.125f;                 // 1/sqrt(64)
    const size_t tokbase = (size_t)bz * SEQ;

    float* Srow = sS[wave]; bf16_t* Pt = sP[wave]; float* Fct = sF[wave];

    // Q A-fragments for the two 32-wide K-steps of the 64-dim head
    const bf16_t* qrow = Q + (tokbase + qt * 16 + r) * D_MODEL + hd * DK;
    v16bf aq0 = cat8(*(const v8bf*)(qrow + 0 + hh * 16),
                     *(const v8bf*)(qrow + 8 + hh * 16));
    v16bf aq1 = cat8(*(const v8bf*)(qrow + 32 + hh * 16),
                     *(const v8bf*)(qrow + 40 + hh * 16));

    v8f oacc[4];
#pragma unroll
    for (int nt = 0; nt < 4; ++nt)
#pragma unroll
        for (int i = 0; i < 8; ++i) oacc[nt][i] = 0.f;

    float m_r = -3.0e38f, l_r = 0.f;   // valid in lanes 0..15 (row owners)

    for (int kb = 0; kb < SEQ / 32; ++kb) {
        // ---- scores: two 16x16 tiles covering keys kb*32 .. kb*32+31 ----
        v8f s0, s1;
#pragma unroll
        for (int i = 0; i < 8; ++i) { s0[i] = 0.f; s1[i] = 0.f; }
#pragma unroll
        for (int ks = 0; ks < 2; ++ks) {
            const bf16_t* k0p = Km + (tokbase + kb * 32 + r) * D_MODEL
                                   + hd * DK + ks * 32 + hh * 16;
            const bf16_t* k1p = Km + (tokbase + kb * 32 + 16 + r) * D_MODEL
                                   + hd * DK + ks * 32 + hh * 16;
            v16bf b0 = *(const v16bf*)k0p;   // contiguous 32B of a K row
            v16bf b1 = *(const v16bf*)k1p;
            v16bf aq = ks ? aq1 : aq0;
            s0 = wmma_bf16(aq, b0, s0);
            s1 = wmma_bf16(aq, b1, s1);
        }
        // ---- preload V fragments (independent of softmax) ----
        v16bf bv[4];
#pragma unroll
        for (int nt = 0; nt < 4; ++nt) {
            const bf16_t* vp = VT + ((size_t)bh * DK + nt * 16 + r) * SEQ
                                  + kb * 32 + hh * 16;
            bv[nt] = *(const v16bf*)vp;      // contiguous 32B of a VT row
        }
        // ---- scatter scores to LDS (D-fragment -> row-major) ----
#pragma unroll
        for (int i = 0; i < 8; ++i) {
            Srow[(hh * 8 + i) * 32 + r]      = s0[i] * scale;
            Srow[(hh * 8 + i) * 32 + 16 + r] = s1[i] * scale;
        }
        asm volatile("s_wait_dscnt 0x0" ::: "memory");
        // ---- online softmax: lanes 0..15 each own one query row ----
        if (lane < 16) {
            const v4f* sr4 = (const v4f*)&Srow[lane * 32];
            float sbuf[32];
#pragma unroll
            for (int q = 0; q < 8; ++q) {
                v4f tq = sr4[q];               // ds_load_b128
                sbuf[4 * q + 0] = tq[0]; sbuf[4 * q + 1] = tq[1];
                sbuf[4 * q + 2] = tq[2]; sbuf[4 * q + 3] = tq[3];
            }
            float tmax = sbuf[0];
#pragma unroll
            for (int j = 1; j < 32; ++j) tmax = fmaxf(tmax, sbuf[j]);
            float m_new = fmaxf(m_r, tmax);
            float f = __expf(m_r - m_new);
            float tsum = 0.f;
            v8bf* pt8 = (v8bf*)&Pt[lane * 32];
#pragma unroll
            for (int g = 0; g < 4; ++g) {
                v8bf pv;
#pragma unroll
                for (int e = 0; e < 8; ++e) {
                    float p = __expf(sbuf[g * 8 + e] - m_new);
                    tsum += p;
                    pv[e] = (bf16_t)p;
                }
                pt8[g] = pv;                   // 16B ds_store
            }
            l_r = l_r * f + tsum;
            m_r = m_new;
            Fct[lane] = f;
        }
        asm volatile("s_wait_dscnt 0x0" ::: "memory");
        // ---- rescale accumulators by per-row factor ----
        float fr[8];
#pragma unroll
        for (int i = 0; i < 8; ++i) fr[i] = Fct[hh * 8 + i];
#pragma unroll
        for (int nt = 0; nt < 4; ++nt)
#pragma unroll
            for (int i = 0; i < 8; ++i) oacc[nt][i] *= fr[i];
        // ---- P A-fragment from LDS, then P @ V over the 32 keys ----
        v16bf ap = cat8(*(const v8bf*)&Pt[r * 32 + hh * 16],
                        *(const v8bf*)&Pt[r * 32 + 8 + hh * 16]);
#pragma unroll
        for (int nt = 0; nt < 4; ++nt)
            oacc[nt] = wmma_bf16(ap, bv[nt], oacc[nt]);
    }
    // ---- final 1/l normalization and store ----
    if (lane < 16) Fct[lane] = 1.f / l_r;
    asm volatile("s_wait_dscnt 0x0" ::: "memory");
    float fr[8];
#pragma unroll
    for (int i = 0; i < 8; ++i) fr[i] = Fct[hh * 8 + i];
#pragma unroll
    for (int nt = 0; nt < 4; ++nt)
#pragma unroll
        for (int i = 0; i < 8; ++i) {
            O[(tokbase + qt * 16 + hh * 8 + i) * D_MODEL
              + hd * DK + nt * 16 + r] = (bf16_t)(oacc[nt][i] * fr[i]);
        }
}

// ---------------------------------------------------------------------------
extern "C" void kernel_launch(void* const* d_in, const int* in_sizes, int n_in,
                              void* d_out, int out_size, void* d_ws, size_t ws_size,
                              hipStream_t stream) {
    (void)in_sizes; (void)n_in; (void)out_size; (void)ws_size;
    const float* x    = (const float*)d_in[0];
    const float* ln1g = (const float*)d_in[1];
    const float* ln1b = (const float*)d_in[2];
    const float* wq   = (const float*)d_in[3];
    const float* bq   = (const float*)d_in[4];
    const float* wk   = (const float*)d_in[5];
    const float* bk   = (const float*)d_in[6];
    const float* wv   = (const float*)d_in[7];
    const float* bv   = (const float*)d_in[8];
    const float* wo   = (const float*)d_in[9];
    const float* bo   = (const float*)d_in[10];
    const float* ln2g = (const float*)d_in[11];
    const float* ln2b = (const float*)d_in[12];
    const float* w1   = (const float*)d_in[13];
    const float* b1   = (const float*)d_in[14];
    const float* w2   = (const float*)d_in[15];
    const float* b2   = (const float*)d_in[16];
    float* out = (float*)d_out;

    char* ws = (char*)d_ws;
    size_t off = 0;
    auto alloc = [&](size_t bytes) -> void* {
        void* p = ws + off;
        off = (off + bytes + 255) & ~(size_t)255;
        return p;
    };
    const size_t ACTB = (size_t)TOK * D_MODEL * sizeof(bf16_t);   // 24 MB

    bf16_t* d_h    = (bf16_t*)alloc(ACTB);                        // ln1 / ln2 out
    bf16_t* d_q    = (bf16_t*)alloc(ACTB);
    bf16_t* d_k    = (bf16_t*)alloc(ACTB);
    bf16_t* d_v    = (bf16_t*)alloc(ACTB);
    bf16_t* d_vt   = (bf16_t*)alloc(ACTB);
    bf16_t* d_attn = d_v;  // V dead after transpose; reuse for attention out
    bf16_t* d_ffn  = (bf16_t*)alloc((size_t)TOK * D_FFN * sizeof(bf16_t));
    bf16_t* wqp = (bf16_t*)alloc((size_t)D_MODEL * D_MODEL * sizeof(bf16_t));
    bf16_t* wkp = (bf16_t*)alloc((size_t)D_MODEL * D_MODEL * sizeof(bf16_t));
    bf16_t* wvp = (bf16_t*)alloc((size_t)D_MODEL * D_MODEL * sizeof(bf16_t));
    bf16_t* wop = (bf16_t*)alloc((size_t)D_MODEL * D_MODEL * sizeof(bf16_t));
    bf16_t* w1p = (bf16_t*)alloc((size_t)D_MODEL * D_FFN * sizeof(bf16_t));
    bf16_t* w2p = (bf16_t*)alloc((size_t)D_FFN * D_MODEL * sizeof(bf16_t));

    // --- pack weights into WMMA B-fragment order (bf16) ---
    const int KN = D_MODEL * D_MODEL;
    pack_w<<<(KN + 255) / 256, 256, 0, stream>>>(wq, wqp, D_MODEL, D_MODEL);
    pack_w<<<(KN + 255) / 256, 256, 0, stream>>>(wk, wkp, D_MODEL, D_MODEL);
    pack_w<<<(KN + 255) / 256, 256, 0, stream>>>(wv, wvp, D_MODEL, D_MODEL);
    pack_w<<<(KN + 255) / 256, 256, 0, stream>>>(wo, wop, D_MODEL, D_MODEL);
    pack_w<<<(D_MODEL * D_FFN + 255) / 256, 256, 0, stream>>>(w1, w1p, D_MODEL, D_FFN);
    pack_w<<<(D_FFN * D_MODEL + 255) / 256, 256, 0, stream>>>(w2, w2p, D_FFN, D_MODEL);

    // --- ln1 -> h (bf16) ---
    layernorm_bf16<<<TOK / 8, 256, 0, stream>>>(x, ln1g, ln1b, d_h, TOK);

    // --- Q, K, V projections ---
    dim3 gqkv(TOK / 128, D_MODEL / 64);
    gemm_bf16_wmma<<<gqkv, 256, 0, stream>>>(d_h, wqp, bq, nullptr, nullptr, d_q,
                                             TOK, D_MODEL, D_MODEL, 0);
    gemm_bf16_wmma<<<gqkv, 256, 0, stream>>>(d_h, wkp, bk, nullptr, nullptr, d_k,
                                             TOK, D_MODEL, D_MODEL, 0);
    gemm_bf16_wmma<<<gqkv, 256, 0, stream>>>(d_h, wvp, bv, nullptr, nullptr, d_v,
                                             TOK, D_MODEL, D_MODEL, 0);

    // --- V transpose for contiguous P@V B-fragments ---
    transpose_v<<<((size_t)TOK * D_MODEL + 255) / 256, 256, 0, stream>>>(d_v, d_vt);

    // --- attention ---
    dim3 ga(SEQ / 16 / 8, BATCH * N_HEADS);
    flash_attn<<<ga, 256, 0, stream>>>(d_q, d_k, d_vt, d_attn);

    // --- O projection + residual -> x2 (fp32 in d_out) ---
    gemm_bf16_wmma<<<gqkv, 256, 0, stream>>>(d_attn, wop, bo, x, out, nullptr,
                                             TOK, D_MODEL, D_MODEL, 0);

    // --- ln2 -> h (bf16, reuse) ---
    layernorm_bf16<<<TOK / 8, 256, 0, stream>>>(out, ln2g, ln2b, d_h, TOK);

    // --- FFN1 + exact GELU -> bf16 ---
    dim3 gf1(TOK / 128, D_FFN / 64);
    gemm_bf16_wmma<<<gf1, 256, 0, stream>>>(d_h, w1p, b1, nullptr, nullptr, d_ffn,
                                            TOK, D_FFN, D_MODEL, 1);

    // --- FFN2 + residual -> final fp32 output ---
    dim3 gf2(TOK / 128, D_MODEL / 64);
    gemm_bf16_wmma<<<gf2, 256, 0, stream>>>(d_ffn, w2p, b2, out, out, nullptr,
                                            TOK, D_MODEL, D_FFN, 0);
}
